// SelfAttention_47201690583124
// MI455X (gfx1250) — matrix operand order repaired and checked by reading, not compiled, not run on hardware
//
#include <hip/hip_runtime.h>
#include <hip/hip_bf16.h>
#include <stdint.h>

// ---------------------------------------------------------------------------
// CDNA5 (gfx1250) wave32 WMMA implementation of the dual-context attention
// block.  All GEMM-shaped math runs on v_wmma_f32_16x16x32_f16; GEMM B-panels
// are staged block-wide into LDS with global_load_async_to_lds_b128
// (ASYNCcnt double-buffered).
// ---------------------------------------------------------------------------

typedef __attribute__((ext_vector_type(16))) _Float16 v16h;
typedef __attribute__((ext_vector_type(8)))  _Float16 h8;
typedef __attribute__((ext_vector_type(8)))  float    v8f;

union AFrag {
  v16h     v;
  _Float16 h[16];
  h8       g[2];
};

__device__ __forceinline__ v8f wmma_f16(const AFrag& a, const AFrag& b, v8f c) {
  // D = A(16x32 f16) * B(32x16 f16) + C(16x16 f32)
  return __builtin_amdgcn_wmma_f32_16x16x32_f16(
      /*neg_a=*/false, a.v, /*neg_b=*/false, b.v,
      /*c_mod=*/(short)0, c, /*reuse_a=*/false, /*reuse_b=*/false);
}

// A-fragment from row-major f16 (p already includes row*ld + (lane>>4)*8):
// element e -> K = (e/8)*16 + (lane/16)*8 + (e%8)
__device__ __forceinline__ void load_a(AFrag& f, const _Float16* p) {
  f.g[0] = *(const h8*)(p);
  f.g[1] = *(const h8*)(p + 16);
}

// B-fragment from packed layout: frag is 512 contiguous halves,
// N = lane&15, 16 halves per lane = K = (e/8)*16 + (lane/16)*8 + (e%8).
__device__ __forceinline__ void load_b(AFrag& f, const _Float16* frag, int lane) {
  const _Float16* p = frag + lane * 16;
  f.g[0] = *(const h8*)(p);
  f.g[1] = *(const h8*)(p + 8);
}

// CDNA5 async global->LDS copy (16B per lane), tracked by ASYNCcnt.
__device__ __forceinline__ void async_g2l_b128(uint32_t lds_byte_addr,
                                               const void* gaddr) {
  asm volatile("global_load_async_to_lds_b128 %0, %1, off"
               :: "v"(lds_byte_addr), "v"((uint64_t)gaddr) : "memory");
}
__device__ __forceinline__ void wait_async0() {
  asm volatile("s_wait_asynccnt 0x0" ::: "memory");
}

__device__ __forceinline__ float block_reduce_sum(float v, float* red, int tid) {
  red[tid] = v;
  __syncthreads();
  for (int st = 128; st > 0; st >>= 1) {
    if (tid < st) red[tid] += red[tid + st];
    __syncthreads();
  }
  float r = red[0];
  __syncthreads();
  return r;
}

// ---------------------------------------------------------------------------
// Elementwise f32 -> f16
// ---------------------------------------------------------------------------
__global__ void f32_to_f16_kernel(const float* __restrict__ in,
                                  _Float16* __restrict__ out, int n) {
  int i = blockIdx.x * blockDim.x + threadIdx.x;
  if (i < n) out[i] = (_Float16)in[i];
}

// ---------------------------------------------------------------------------
// Pack a row-major f32 weight W[N][K] into WMMA B-fragments:
// out[((n/16)*(K/32) + k/32)*512 + ((n%16) + 16*((k>>3)&1))*16 + ((k>>4)&1)*8 + (k&7)] = W[n][k]
// ---------------------------------------------------------------------------
__global__ void pack_wB_kernel(const float* __restrict__ W,
                               _Float16* __restrict__ out, int N, int K) {
  int idx = blockIdx.x * blockDim.x + threadIdx.x;
  if (idx >= N * K) return;
  int frag = idx >> 9;
  int r    = idx & 511;
  int lane = r >> 4;   // N = lane&15, K half = lane>>4
  int e    = r & 15;   // K = (e>>3)*16 + (lane>>4)*8 + (e&7)
  int KT   = K >> 5;
  int nt   = frag / KT;
  int kt   = frag - nt * KT;
  out[idx] = (_Float16)W[(size_t)(nt * 16 + (lane & 15)) * K + kt * 32 + (e >> 3) * 16 + (lane >> 4) * 8 + (e & 7)];
}

// ---------------------------------------------------------------------------
// GEMM:  C[M][N] = A[M][K](f16, row-major) * Bp(packed f16) + bias
// Block = 8 waves = 256x64 macro tile; wave = 32x64 tile (8 WMMA / k-step).
// B panel (4KB / k-step) staged into double-buffered LDS with
// global_load_async_to_lds_b128: each of the 8 waves issues exactly one
// async b128 per k-step (8 x 512B = 4KB), then s_wait_asynccnt + barrier.
// ---------------------------------------------------------------------------
__global__ void gemm_wave_kernel(const _Float16* __restrict__ A,
                                 const _Float16* __restrict__ Bp,
                                 float* __restrict__ C,
                                 const float* __restrict__ bias,
                                 int M, int N, int K) {
  __shared__ _Float16 bsm[2][4][512];   // 8KB double-buffered B panel

  const int tid  = threadIdx.x;
  const int lane = tid & 31;
  const int w    = tid >> 5;            // wave 0..7
  const int numN = N >> 6;
  const int bM   = blockIdx.x / numN;
  const int nb   = blockIdx.x - bM * numN;
  const int n0   = nb << 6;
  const int m0   = bM * 256 + w * 32;
  const int KT   = K >> 5;
  const int mrow = lane & 15;
  const int hi   = lane >> 4;
  const int cl   = lane & 15;

  // This thread's role in the cooperative B stage: wave w covers
  // fragment j = w/2, half h = w%2; lane moves 16 bytes.
  const int jj = w >> 1;
  const int hf = w & 1;
  const _Float16* bsrc =
      Bp + (size_t)((n0 >> 4) + jj) * KT * 512 + hf * 256 + lane * 8;

  const _Float16* arow0 = A + (size_t)(m0 + mrow) * K + hi * 8;
  const _Float16* arow1 = arow0 + (size_t)16 * K;

  v8f acc[8];
  v8f zero = {};
#pragma unroll
  for (int i = 0; i < 8; ++i) acc[i] = zero;

  // Prologue: stage k-step 0 into buffer 0
  async_g2l_b128((uint32_t)(uintptr_t)&bsm[0][jj][hf * 256 + lane * 8], bsrc);
  wait_async0();
  __syncthreads();

  for (int kt = 0; kt < KT; ++kt) {
    const int cur = kt & 1;
    if (kt + 1 < KT) {
      async_g2l_b128((uint32_t)(uintptr_t)&bsm[cur ^ 1][jj][hf * 256 + lane * 8],
                     bsrc + (size_t)(kt + 1) * 512);
    }
    AFrag a0, a1;
    load_a(a0, arow0 + kt * 32);
    load_a(a1, arow1 + kt * 32);
    __builtin_prefetch(arow0 + kt * 32 + 64, 0, 1);
    __builtin_prefetch(arow1 + kt * 32 + 64, 0, 1);
#pragma unroll
    for (int j = 0; j < 4; ++j) {
      AFrag b;
      const _Float16* bp = &bsm[cur][j][lane * 16];
      b.g[0] = *(const h8*)(bp);       // ds_load_b128
      b.g[1] = *(const h8*)(bp + 8);
      acc[2 * j]     = wmma_f16(a0, b, acc[2 * j]);
      acc[2 * j + 1] = wmma_f16(a1, b, acc[2 * j + 1]);
    }
    wait_async0();       // next buffer landed
    __syncthreads();     // everyone done reading current buffer
  }

#pragma unroll
  for (int j = 0; j < 4; ++j) {
    const float bb = bias ? bias[n0 + j * 16 + cl] : 0.f;
#pragma unroll
    for (int r = 0; r < 2; ++r) {
#pragma unroll
      for (int v = 0; v < 8; ++v) {
        const int row = m0 + r * 16 + v + 8 * hi;  // C layout: M = v + 8*(lane/16)
        C[(size_t)row * N + n0 + j * 16 + cl] = acc[2 * j + r][v] + bb;
      }
    }
  }
}

// ---------------------------------------------------------------------------
// Fused RMSNorm + RoPE + layout build for Q/K/V.
// comb: [2 ctx][2048 rows][1536] f32  (row = b*1024 + s)
//   [0,1024)    q   (rmsnorm over 1024, then rope per 64-head)
//   [1024,1280) k   (rmsnorm over 256, then rope)
//   [1280,1536) v
// Outputs:
//   qh : f16 row-major [b*16+h][1024 s][128 d]   (d = ctx*64 + d0)
//   kp : packed B-frags for S=QK^T :
//        [((b*4+kv)*4 + d/32)*64 + s/16]*512 + ((s%16) + 16*((d>>3)&1))*16 + ((d>>4)&1)*8 + (d&7)
//   vp : packed B-frags for P*V :
//        [((b*4+kv)*32 + s/32)*8 + d/16]*512 + ((d%16) + 16*((s>>3)&1))*16 + ((s>>4)&1)*8 + (s&7)
// ---------------------------------------------------------------------------
__global__ void build_qkv_kernel(const float* __restrict__ comb,
                                 const float* __restrict__ fcos,
                                 const float* __restrict__ fsin,
                                 const float* __restrict__ gql,
                                 const float* __restrict__ gkl,
                                 const float* __restrict__ gqg,
                                 const float* __restrict__ gkg,
                                 _Float16* __restrict__ qh,
                                 _Float16* __restrict__ kp,
                                 _Float16* __restrict__ vp) {
  __shared__ float red[256];
  const int m   = blockIdx.x;      // b*1024 + s
  const int b   = m >> 10;
  const int s   = m & 1023;
  const int tid = threadIdx.x;

  for (int ctx = 0; ctx < 2; ++ctx) {
    const float* row = comb + ((size_t)ctx * 2048 + m) * 1536;
    const float* gq  = ctx ? gqg : gql;
    const float* gk  = ctx ? gkg : gkl;

    float pq = 0.f;
    for (int i = tid; i < 1024; i += 256) { float x = row[i]; pq += x * x; }
    float ssq_q = block_reduce_sum(pq, red, tid);

    float xk = row[1024 + tid];
    float ssq_k = block_reduce_sum(xk * xk, red, tid);

    const float rq = rsqrtf(ssq_q / 1024.f + 1e-5f);
    const float rk = rsqrtf(ssq_k / 256.f  + 1e-5f);

    // Q: 512 rope pairs
    for (int p = tid; p < 512; p += 256) {
      const int i0 = 2 * p;
      const int hh = i0 >> 6;
      const int d0 = i0 & 63;
      const int j  = d0 >> 1;
      const float c  = fcos[s * 32 + j];
      const float sn = fsin[s * 32 + j];
      const float x0 = row[i0]     * rq * gq[i0];
      const float x1 = row[i0 + 1] * rq * gq[i0 + 1];
      const float o0 = x0 * c - x1 * sn;
      const float o1 = x0 * sn + x1 * c;
      const size_t base = ((size_t)(b * 16 + hh) * 1024 + s) * 128 + ctx * 64 + d0;
      qh[base]     = (_Float16)o0;
      qh[base + 1] = (_Float16)o1;
    }

    // K: 128 rope pairs -> packed B-frag for QK^T
    for (int p = tid; p < 128; p += 256) {
      const int i0  = 2 * p;            // 0..255
      const int kvh = i0 >> 6;
      const int d0  = i0 & 63;
      const int j   = d0 >> 1;
      const float c  = fcos[s * 32 + j];
      const float sn = fsin[s * 32 + j];
      const float x0 = row[1024 + i0]     * rk * gk[i0];
      const float x1 = row[1024 + i0 + 1] * rk * gk[i0 + 1];
      const float o0 = x0 * c - x1 * sn;
      const float o1 = x0 * sn + x1 * c;
      const int d = ctx * 64 + d0;
      const size_t i0k = ((size_t)((b * 4 + kvh) * 4 + (d >> 5)) * 64 + (s >> 4)) * 512
                         + (((d >> 3) & 1) * 16 + (s & 15)) * 16 + ((d >> 4) & 1) * 8 + (d & 7);
      const int dn = d + 1;
      const size_t i1k = ((size_t)((b * 4 + kvh) * 4 + (dn >> 5)) * 64 + (s >> 4)) * 512
                         + (((dn >> 3) & 1) * 16 + (s & 15)) * 16 + ((dn >> 4) & 1) * 8 + (dn & 7);
      kp[i0k] = (_Float16)o0;
      kp[i1k] = (_Float16)o1;
    }

    // V: 256 elements -> packed B-frag for P*V
    {
      const int i   = tid;              // 0..255
      const int kvh = i >> 6;
      const int d   = ctx * 64 + (i & 63);
      const float x = row[1280 + i];
      const size_t iv = ((size_t)((b * 4 + kvh) * 32 + (s >> 5)) * 8 + (d >> 4)) * 512
                        + (((s >> 3) & 1) * 16 + (d & 15)) * 16 + ((s >> 4) & 1) * 8 + (s & 7);
      vp[iv] = (_Float16)x;
    }
    __syncthreads();
  }
}

// ---------------------------------------------------------------------------
// Attention: one block = 4 waves = 64 q rows for one (b,h).
// Phase 1: S = (Q K^T)/8 via WMMA into 256KB LDS tile [64][1024].
// Phase 2: exact row softmax in LDS.
// Phase 3: O = P V via WMMA; store f16 row-major out[m][2048], col = h*128+d.
// ---------------------------------------------------------------------------
__global__ void attn_kernel(const _Float16* __restrict__ qh,
                            const _Float16* __restrict__ kp,
                            const _Float16* __restrict__ vp,
                            _Float16* __restrict__ outh) {
  extern __shared__ char smem[];
  float* sc   = (float*)smem;                              // [64][1024]
  float* rinv = (float*)(smem + (size_t)64 * 1024 * 4);    // [64]

  const int lane = threadIdx.x & 31;
  const int w    = threadIdx.x >> 5;          // wave 0..3
  const int qt   = blockIdx.x & 15;
  const int bh   = blockIdx.x >> 4;
  const int h    = bh & 15;
  const int b    = bh >> 4;
  const int kv   = h >> 2;                    // GQA: 4 q-heads per kv-head
  const int sq0  = qt * 64 + w * 16;
  const int mrow = lane & 15;
  const int hi   = lane >> 4;
  const int cl   = lane & 15;

  // Hoist the 4 Q fragments (d = 0..127 in chunks of 32)
  const _Float16* qbase =
      qh + ((size_t)(b * 16 + h) * 1024 + sq0 + mrow) * 128 + hi * 8;
  AFrag qa0, qa1, qa2, qa3;
  load_a(qa0, qbase + 0);
  load_a(qa1, qbase + 32);
  load_a(qa2, qbase + 64);
  load_a(qa3, qbase + 96);

  const _Float16* kbase = kp + (size_t)(b * 4 + kv) * 4 * 64 * 512;
  for (int skt = 0; skt < 64; ++skt) {
    v8f acc = {};
    AFrag kb;
    load_b(kb, kbase + ((size_t)0 * 64 + skt) * 512, lane); acc = wmma_f16(qa0, kb, acc);
    load_b(kb, kbase + ((size_t)1 * 64 + skt) * 512, lane); acc = wmma_f16(qa1, kb, acc);
    load_b(kb, kbase + ((size_t)2 * 64 + skt) * 512, lane); acc = wmma_f16(qa2, kb, acc);
    load_b(kb, kbase + ((size_t)3 * 64 + skt) * 512, lane); acc = wmma_f16(qa3, kb, acc);
#pragma unroll
    for (int v = 0; v < 8; ++v)
      sc[(size_t)(w * 16 + v + 8 * hi) * 1024 + skt * 16 + cl] = acc[v] * 0.125f;
  }
  __syncthreads();

  // Exact softmax: one thread per row
  if (threadIdx.x < 64) {
    float* row = sc + (size_t)threadIdx.x * 1024;
    float mx = -1e30f;
    for (int i = 0; i < 1024; ++i) mx = fmaxf(mx, row[i]);
    float sum = 0.f;
    for (int i = 0; i < 1024; ++i) {
      float e = __expf(row[i] - mx);
      row[i] = e;
      sum += e;
    }
    rinv[threadIdx.x] = 1.f / sum;
  }
  __syncthreads();

  v8f zero = {};
  v8f o[8];
#pragma unroll
  for (int i = 0; i < 8; ++i) o[i] = zero;

  const int prow = w * 16 + mrow;        // A-frag row for this lane
  const float ri = rinv[prow];
  const float* srow = sc + (size_t)prow * 1024;
  const _Float16* vbase = vp + (size_t)(b * 4 + kv) * 32 * 8 * 512;

  for (int skc = 0; skc < 32; ++skc) {
    AFrag pf;
    const int kb0 = skc * 32 + hi * 8;
#pragma unroll
    for (int e = 0; e < 16; ++e) {
      const int kk = kb0 + ((e >> 3) << 4) + (e & 7);
      pf.h[e] = (_Float16)(srow[kk] * ri);
    }
#pragma unroll
    for (int dt = 0; dt < 8; ++dt) {
      AFrag vf;
      load_b(vf, vbase + ((size_t)skc * 8 + dt) * 512, lane);
      o[dt] = wmma_f16(pf, vf, o[dt]);
    }
  }

#pragma unroll
  for (int dt = 0; dt < 8; ++dt) {
#pragma unroll
    for (int v = 0; v < 8; ++v) {
      const int sq = sq0 + v + 8 * hi;
      const size_t mrow_g = (size_t)b * 1024 + sq;
      outh[mrow_g * 2048 + h * 128 + dt * 16 + cl] = (_Float16)o[dt][v];
    }
  }
}

// ---------------------------------------------------------------------------
// Final RMSNorm of both output streams into d_out (x_lc then x_gc, fp32)
// ---------------------------------------------------------------------------
__global__ void final_rms_kernel(const float* __restrict__ y,
                                 const float* __restrict__ gl,
                                 const float* __restrict__ gg,
                                 float* __restrict__ out) {
  __shared__ float red[256];
  const int m   = blockIdx.x;   // 0..2047
  const int tid = threadIdx.x;
  for (int o = 0; o < 2; ++o) {
    const float* row = y + ((size_t)o * 2048 + m) * 1024;
    const float* g   = o ? gg : gl;
    float p = 0.f;
    for (int i = tid; i < 1024; i += 256) { float x = row[i]; p += x * x; }
    float ssq = block_reduce_sum(p, red, tid);
    const float r = rsqrtf(ssq / 1024.f + 1e-5f);
    for (int i = tid; i < 1024; i += 256)
      out[(size_t)o * 2048 * 1024 + (size_t)m * 1024 + i] = row[i] * r * g[i];
    __syncthreads();
  }
}

// ---------------------------------------------------------------------------
// Host-side orchestration
// ---------------------------------------------------------------------------
extern "C" void kernel_launch(void* const* d_in, const int* in_sizes, int n_in,
                              void* d_out, int out_size, void* d_ws, size_t ws_size,
                              hipStream_t stream) {
  (void)in_sizes; (void)n_in; (void)out_size; (void)ws_size;

  const float* local_c   = (const float*)d_in[0];
  const float* global_c  = (const float*)d_in[1];
  const float* freqs_cos = (const float*)d_in[2];
  const float* freqs_sin = (const float*)d_in[3];
  const float* W_lc      = (const float*)d_in[4];
  const float* W_gc      = (const float*)d_in[5];
  const float* g_q_lc    = (const float*)d_in[6];
  const float* g_k_lc    = (const float*)d_in[7];
  const float* g_q_gc    = (const float*)d_in[8];
  const float* g_k_gc    = (const float*)d_in[9];
  const float* W_local   = (const float*)d_in[10];
  const float* b_local   = (const float*)d_in[11];
  const float* g_lc_out  = (const float*)d_in[12];
  const float* W_global  = (const float*)d_in[13];
  const float* b_global  = (const float*)d_in[14];
  const float* g_gc_out  = (const float*)d_in[15];

  char* ws = (char*)d_ws;
  // Fixed 256B-aligned workspace layout (total 85,983,232 B)
  _Float16* Xh   = (_Float16*)(ws + 0);          // [2][2048][1024] f16
  _Float16* Wqkv = (_Float16*)(ws + 8388608);    // [2] packed 1536x1024
  float*    comb = (float*)   (ws + 14680064);   // [2][2048][1536] f32
  _Float16* qh   = (_Float16*)(ws + 39845888);   // [2*16][1024][128] f16
  _Float16* kp   = (_Float16*)(ws + 48234496);   // packed K frags
  _Float16* vp   = (_Float16*)(ws + 50331648);   // packed V frags
  _Float16* outh = (_Float16*)(ws + 52428800);   // [2048][2048] f16
  _Float16* Wout = (_Float16*)(ws + 60817408);   // [2] packed 1024x2048
  float*    y    = (float*)   (ws + 69206016);   // [2][2048][1024] f32

  // 1) fp32 -> fp16 activations
  f32_to_f16_kernel<<<8192, 256, 0, stream>>>(local_c,  Xh,            2097152);
  f32_to_f16_kernel<<<8192, 256, 0, stream>>>(global_c, Xh + 2097152,  2097152);

  // 2) pack QKV weights into B-fragments
  pack_wB_kernel<<<6144, 256, 0, stream>>>(W_lc, Wqkv,            1536, 1024);
  pack_wB_kernel<<<6144, 256, 0, stream>>>(W_gc, Wqkv + 1572864,  1536, 1024);

  // 3) QKV projection GEMMs: comb[ctx] = X[ctx] @ W[ctx]^T  (2048x1536x1024)
  //    grid = (2048/256) * (1536/64) = 192 blocks of 8 waves
  gemm_wave_kernel<<<192, 256, 0, stream>>>(Xh,           Wqkv,           comb,
                                            nullptr, 2048, 1536, 1024);
  gemm_wave_kernel<<<192, 256, 0, stream>>>(Xh + 2097152, Wqkv + 1572864, comb + 3145728,
                                            nullptr, 2048, 1536, 1024);

  // 4) fused rmsnorm + rope + Q/K/V fragment build
  build_qkv_kernel<<<2048, 256, 0, stream>>>(comb, freqs_cos, freqs_sin,
                                             g_q_lc, g_k_lc, g_q_gc, g_k_gc,
                                             qh, kp, vp);

  // 5) attention (dynamic LDS: 64*1024*4 + 64*4 = 262400 B, < 320KB/WG)
  attn_kernel<<<512, 128, 262400, stream>>>(qh, kp, vp, outh);

  // 6) pack output weights
  pack_wB_kernel<<<8192, 256, 0, stream>>>(W_local,  Wout,           1024, 2048);
  pack_wB_kernel<<<8192, 256, 0, stream>>>(W_global, Wout + 2097152, 1024, 2048);

  // 7) output GEMMs with bias: y[o] = out @ W^T + b  (2048x1024x2048)
  //    grid = (2048/256) * (1024/64) = 128 blocks
  gemm_wave_kernel<<<128, 256, 0, stream>>>(outh, Wout,           y,
                                            b_local,  2048, 1024, 2048);
  gemm_wave_kernel<<<128, 256, 0, stream>>>(outh, Wout + 2097152, y + 2097152,
                                            b_global, 2048, 1024, 2048);

  // 8) final rmsnorm into d_out (x_lc ‖ x_gc)
  final_rms_kernel<<<2048, 256, 0, stream>>>(y, g_lc_out, g_gc_out, (float*)d_out);
}
